// MultiHeadAttention_89867895702150
// MI455X (gfx1250) — matrix operand order repaired and checked by reading, not compiled, not run on hardware
//
#include <hip/hip_runtime.h>
#include <hip/hip_bf16.h>
#include <stdint.h>

// ---------------------------------------------------------------------------
// MultiHeadAttention for MI455X (gfx1250, wave32, WMMA, async-to-LDS)
//   x:   [4, 2048, 1024] f32
//   W_*: [16, 64, 1024]  f32   (torch Linear layout: [out, in])
//   out: [4, 2048, 1024] f32   (heads concatenated)
// Pipeline:
//   1) qkv_proj: f32->f16 convert + WMMA GEMM -> Q/K/V f16 planes in d_ws
//      (double-buffered LDS tiles, global loads overlapped with WMMA)
//   2) attn: flash-attention per (b*h, 128-row q tile) with WMMA f16/f32-acc
//      (Q/K tiles staged with global_load_async_to_lds_b128, double-buffered
//       K/V, s_wait_asynccnt pipeline)
// ---------------------------------------------------------------------------

typedef __attribute__((ext_vector_type(16))) _Float16 v16h;
typedef __attribute__((ext_vector_type(8)))  float    v8f;

#define D_IN     1024
#define D_OUT    64
#define HEADS    16
#define SEQ      2048
#define BATCH    4
#define ROWS     (BATCH * SEQ)          // 8192
#define LDH      72                     // padded leading dim (halves) for LDS tiles
#define PLANE    ((size_t)ROWS * HEADS * D_OUT)   // elements per Q/K/V plane

// ---- CDNA5 async memory->LDS copy (16B per lane, ASYNCcnt-tracked) --------
__device__ __forceinline__ void async_copy_b128(const _Float16* gbase,
                                                uint32_t voff_bytes,
                                                _Float16* lds_dst) {
    const uint32_t ldsa  = (uint32_t)(uintptr_t)lds_dst;   // low 32b = LDS offset
    const uint64_t sbase = (uint64_t)(uintptr_t)gbase;
    asm volatile("global_load_async_to_lds_b128 %0, %1, %2"
                 :: "v"(ldsa), "v"(voff_bytes), "s"(sbase)
                 : "memory");
}
__device__ __forceinline__ void wait_async0() {
    asm volatile("s_wait_asynccnt 0" ::: "memory");
}

// ---- WMMA fragment loaders (CDNA5 16x16x32 f16, wave32) -------------------
// A (16x32, MxK): lanes 0-15 hold M=lane, K 0..7 in v0-3 + K 16..23 in v4-7;
//                 lanes 16-31 hold M=lane-16 with K offset +8.
__device__ __forceinline__ v16h load_a16x32(const _Float16* base, int ld, int lane) {
    union { uint32_t u[8]; v16h h; } f;
    const int row = lane & 15;
    const int sel = (lane >> 4) * 8;
    const _Float16* p = base + row * ld;
#pragma unroll
    for (int r = 0; r < 4; ++r)
        f.u[r] = *(const uint32_t*)(p + sel + 2 * r);
#pragma unroll
    for (int r = 0; r < 4; ++r)
        f.u[4 + r] = *(const uint32_t*)(p + 16 + sel + 2 * r);
    return f.h;
}

// B (32x16, KxN) from a column-major staging buffer Bt[n][k]:
// lanes 0-15 hold column N=lane with K 0..7 + K 16..23; lanes 16-31 same column, K offset +8.
__device__ __forceinline__ v16h load_b32x16(const _Float16* base, int ld, int lane) {
    union { uint32_t u[8]; v16h h; } f;
    const int col = lane & 15;
    const int sel = (lane >> 4) * 8;
    const _Float16* p = base + col * ld + sel;
#pragma unroll
    for (int r = 0; r < 8; ++r)
        f.u[r] = *(const uint32_t*)(p + (r >> 2) * 16 + 2 * (r & 3));
    return f.h;
}

__device__ __forceinline__ v8f wmma_f16(v16h a, v16h b, v8f c) {
    return __builtin_amdgcn_wmma_f32_16x16x32_f16(false, a, false, b, (short)0, c,
                                                  false, false);
}

// ---------------------------------------------------------------------------
// Kernel 1: per-head QKV projection.  out[h] = x @ W[h]^T  (f16 result)
// grid = (ROWS/128, HEADS, 3), block = 256 (8 waves), double-buffered LDS
// ---------------------------------------------------------------------------
__global__ __launch_bounds__(256) void qkv_proj_kernel(
    const float* __restrict__ x,
    const float* __restrict__ Wq,
    const float* __restrict__ Wk,
    const float* __restrict__ Wv,
    _Float16* __restrict__ qkv)
{
    __shared__ _Float16 As[2][128 * LDH];   // x tile, f16, row-major [row][k]
    __shared__ _Float16 Bs[2][64 * LDH];    // W tile, f16, [out][k]

    const int mtile = blockIdx.x;
    const int head  = blockIdx.y;
    const int mat   = blockIdx.z;
    const float* W = (mat == 0) ? Wq : (mat == 1) ? Wk : Wv;
    W += (size_t)head * D_OUT * D_IN;
    _Float16* out = qkv + (size_t)mat * PLANE;

    const int tid  = threadIdx.x;
    const int lane = tid & 31;
    const int wave = tid >> 5;
    const int rbase = wave * 16;

    // per-thread staging coordinates
    const int xr_row = tid >> 4;            // 0..15 (x tile row group base)
    const int xr_col = (tid & 15) << 2;     // 0..60

    v8f acc[4];
#pragma unroll
    for (int n = 0; n < 4; ++n)
#pragma unroll
        for (int e = 0; e < 8; ++e) acc[n][e] = 0.0f;

    // ---- prologue: stage chunk 0 into buffer 0 ----
    {
#pragma unroll
        for (int j = 0; j < 8; ++j) {
            const int r = xr_row + 16 * j;
            const float4 v = *(const float4*)(x + (size_t)(mtile * 128 + r) * D_IN + xr_col);
            union { _Float16 h[4]; uint32_t u[2]; } pk;
            pk.h[0] = (_Float16)v.x; pk.h[1] = (_Float16)v.y;
            pk.h[2] = (_Float16)v.z; pk.h[3] = (_Float16)v.w;
            *(uint32_t*)(&As[0][r * LDH + xr_col])     = pk.u[0];
            *(uint32_t*)(&As[0][r * LDH + xr_col + 2]) = pk.u[1];
        }
#pragma unroll
        for (int j = 0; j < 4; ++j) {
            const int r = xr_row + 16 * j;
            const float4 v = *(const float4*)(W + (size_t)r * D_IN + xr_col);
            union { _Float16 h[4]; uint32_t u[2]; } pk;
            pk.h[0] = (_Float16)v.x; pk.h[1] = (_Float16)v.y;
            pk.h[2] = (_Float16)v.z; pk.h[3] = (_Float16)v.w;
            *(uint32_t*)(&Bs[0][r * LDH + xr_col])     = pk.u[0];
            *(uint32_t*)(&Bs[0][r * LDH + xr_col + 2]) = pk.u[1];
        }
    }
    __syncthreads();

    int cur = 0;
    for (int kb = 0; kb < D_IN; kb += 64) {
        const int  nxt     = cur ^ 1;
        const bool hasNext = (kb + 64 < D_IN);

        // issue next chunk's global loads (in flight during WMMA below)
        float4 xr[8], wr[4];
        if (hasNext) {
#pragma unroll
            for (int j = 0; j < 8; ++j) {
                const int r = xr_row + 16 * j;
                xr[j] = *(const float4*)(x + (size_t)(mtile * 128 + r) * D_IN
                                           + (kb + 64) + xr_col);
            }
#pragma unroll
            for (int j = 0; j < 4; ++j) {
                const int r = xr_row + 16 * j;
                wr[j] = *(const float4*)(W + (size_t)r * D_IN + (kb + 64) + xr_col);
            }
        }

        // compute on current buffer
#pragma unroll
        for (int kk = 0; kk < 64; kk += 32) {
            const v16h a = load_a16x32(&As[cur][rbase * LDH + kk], LDH, lane);
#pragma unroll
            for (int n = 0; n < 4; ++n) {
                const v16h b = load_b32x16(&Bs[cur][(n * 16) * LDH + kk], LDH, lane);
                acc[n] = wmma_f16(a, b, acc[n]);
            }
        }

        // convert + store next chunk into the other buffer (no conflict w/ cur)
        if (hasNext) {
#pragma unroll
            for (int j = 0; j < 8; ++j) {
                const int r = xr_row + 16 * j;
                union { _Float16 h[4]; uint32_t u[2]; } pk;
                pk.h[0] = (_Float16)xr[j].x; pk.h[1] = (_Float16)xr[j].y;
                pk.h[2] = (_Float16)xr[j].z; pk.h[3] = (_Float16)xr[j].w;
                *(uint32_t*)(&As[nxt][r * LDH + xr_col])     = pk.u[0];
                *(uint32_t*)(&As[nxt][r * LDH + xr_col + 2]) = pk.u[1];
            }
#pragma unroll
            for (int j = 0; j < 4; ++j) {
                const int r = xr_row + 16 * j;
                union { _Float16 h[4]; uint32_t u[2]; } pk;
                pk.h[0] = (_Float16)wr[j].x; pk.h[1] = (_Float16)wr[j].y;
                pk.h[2] = (_Float16)wr[j].z; pk.h[3] = (_Float16)wr[j].w;
                *(uint32_t*)(&Bs[nxt][r * LDH + xr_col])     = pk.u[0];
                *(uint32_t*)(&Bs[nxt][r * LDH + xr_col + 2]) = pk.u[1];
            }
        }
        __syncthreads();
        cur = nxt;
    }

    // store f16 results into [b][h][s][64] plane
    const int half = lane >> 4;
    const int ln   = lane & 15;
#pragma unroll
    for (int n = 0; n < 4; ++n)
#pragma unroll
        for (int r = 0; r < 8; ++r) {
            const int rowg = mtile * 128 + rbase + half * 8 + r;
            const int b    = rowg >> 11;           // /2048
            const int srow = rowg & 2047;
            const size_t off = (((size_t)(b * HEADS + head) * SEQ + srow) * D_OUT)
                             + n * 16 + ln;
            out[off] = (_Float16)acc[n][r];
        }
}

// ---------------------------------------------------------------------------
// Kernel 2: flash attention over f16 Q/K/V planes, f32 output (concat heads).
// grid = (SEQ/128, BATCH*HEADS), block = 256 (8 waves). K tiles staged with
// async-to-LDS copies, K/V double-buffered under the WMMA compute.
// ---------------------------------------------------------------------------
__global__ __launch_bounds__(256) void attn_kernel(
    const _Float16* __restrict__ qkv,
    float* __restrict__ out)
{
    __shared__ _Float16 Qs[128 * LDH];           // Q tile [qrow][d]
    __shared__ _Float16 Kt[2][64 * LDH];         // K tile [key][d], double buffered
    __shared__ _Float16 Vt[2][64 * LDH];         // V tile transposed [d][key]
    __shared__ _Float16 Ps[8 * 16 * LDH];        // wave-private P staging [qrow][key]

    const int qtile = blockIdx.x;
    const int bh    = blockIdx.y;
    const int tid   = threadIdx.x;
    const int lane  = tid & 31;
    const int wave  = tid >> 5;

    const _Float16* Qh = qkv + 0 * PLANE + (size_t)bh * SEQ * D_OUT;
    const _Float16* Kh = qkv + 1 * PLANE + (size_t)bh * SEQ * D_OUT;
    const _Float16* Vh = qkv + 2 * PLANE + (size_t)bh * SEQ * D_OUT;

    // ---- stage Q tile via async copies (128x64 halves, 16B per lane x4) ----
#pragma unroll
    for (int j = 0; j < 4; ++j) {
        const int i = tid + 256 * j;
        const int r = i >> 3;                    // row in tile
        const int c = (i & 7) << 3;              // halves
        async_copy_b128(Qh, (uint32_t)(((qtile * 128 + r) * D_OUT + c) * 2),
                        &Qs[r * LDH + c]);
    }

    // ---- prologue: stage key tile 0 ----
    {
#pragma unroll
        for (int j = 0; j < 2; ++j) {
            const int i = tid + 256 * j;
            const int r = i >> 3;
            const int c = (i & 7) << 3;
            async_copy_b128(Kh, (uint32_t)((r * D_OUT + c) * 2), &Kt[0][r * LDH + c]);
        }
        union { uint4 v; _Float16 h[8]; } u[2];
#pragma unroll
        for (int j = 0; j < 2; ++j) {
            const int i = tid + 256 * j;
            const int r = i >> 3;
            const int c = (i & 7) << 3;
            u[j].v = *(const uint4*)(Vh + (size_t)r * D_OUT + c);
        }
#pragma unroll
        for (int j = 0; j < 2; ++j) {
            const int i = tid + 256 * j;
            const int r = i >> 3;
            const int c = (i & 7) << 3;
#pragma unroll
            for (int e = 0; e < 8; ++e) Vt[0][(c + e) * LDH + r] = u[j].h[e];
        }
    }
    wait_async0();
    __syncthreads();

    // per-wave persistent Q fragments (d = 0..31, 32..63)
    v16h qfrag[2];
    qfrag[0] = load_a16x32(&Qs[(wave * 16) * LDH + 0],  LDH, lane);
    qfrag[1] = load_a16x32(&Qs[(wave * 16) * LDH + 32], LDH, lane);

    // online-softmax state: 8 rows per lane-half (uniform across 16 lanes)
    float m[8], l[8];
    v8f O[4];
#pragma unroll
    for (int r = 0; r < 8; ++r) { m[r] = -3.0e38f; l[r] = 0.0f; }
#pragma unroll
    for (int n = 0; n < 4; ++n)
#pragma unroll
        for (int e = 0; e < 8; ++e) O[n][e] = 0.0f;

    const int half = lane >> 4;
    const int ln   = lane & 15;
    _Float16* Pw = &Ps[wave * 16 * LDH];
    const float scale = 0.125f;                  // 1/sqrt(64)

    int cur = 0;
    for (int kt = 0; kt < SEQ / 64; ++kt) {
        const int  nxt     = cur ^ 1;
        const bool hasNext = (kt + 1 < SEQ / 64);

        // ---- stage next key tile: async K copy + V loads to regs ----
        union { uint4 v; _Float16 h[8]; } vr[2];
        if (hasNext) {
            const _Float16* Kn = Kh + (size_t)(kt + 1) * 64 * D_OUT;
            const _Float16* Vn = Vh + (size_t)(kt + 1) * 64 * D_OUT;
#pragma unroll
            for (int j = 0; j < 2; ++j) {
                const int i = tid + 256 * j;
                const int r = i >> 3;
                const int c = (i & 7) << 3;
                async_copy_b128(Kn, (uint32_t)((r * D_OUT + c) * 2),
                                &Kt[nxt][r * LDH + c]);
            }
#pragma unroll
            for (int j = 0; j < 2; ++j) {
                const int i = tid + 256 * j;
                const int r = i >> 3;
                const int c = (i & 7) << 3;
                vr[j].v = *(const uint4*)(Vn + (size_t)r * D_OUT + c);
            }
        }

        // ---- S = (Q K^T) * scale : 16 q-rows x 64 keys per wave ----
        v8f S[4];
#pragma unroll
        for (int n = 0; n < 4; ++n) {
#pragma unroll
            for (int e = 0; e < 8; ++e) S[n][e] = 0.0f;
#pragma unroll
            for (int kk = 0; kk < 2; ++kk) {
                const v16h b = load_b32x16(&Kt[cur][(n * 16) * LDH + kk * 32], LDH, lane);
                S[n] = wmma_f16(qfrag[kk], b, S[n]);
            }
        }

        // ---- online softmax ----
        float mnew[8], cf[8], t[8];
#pragma unroll
        for (int r = 0; r < 8; ++r) {
            float v = S[0][r] * scale;
#pragma unroll
            for (int n = 1; n < 4; ++n) v = fmaxf(v, S[n][r] * scale);
#pragma unroll
            for (int sh = 1; sh < 16; sh <<= 1)
                v = fmaxf(v, __shfl_xor(v, sh, 32));
            mnew[r] = fmaxf(m[r], v);
            cf[r]   = __expf(m[r] - mnew[r]);
            m[r]    = mnew[r];
        }
#pragma unroll
        for (int n = 0; n < 4; ++n)
#pragma unroll
            for (int r = 0; r < 8; ++r)
                S[n][r] = __expf(S[n][r] * scale - mnew[r]);
#pragma unroll
        for (int r = 0; r < 8; ++r) {
            float v = S[0][r] + S[1][r] + S[2][r] + S[3][r];
#pragma unroll
            for (int sh = 1; sh < 16; sh <<= 1)
                v += __shfl_xor(v, sh, 32);
            t[r] = v;
            l[r] = l[r] * cf[r] + t[r];
        }
#pragma unroll
        for (int n = 0; n < 4; ++n)
#pragma unroll
            for (int r = 0; r < 8; ++r) O[n][r] *= cf[r];

        // ---- P (C layout) -> wave-private LDS -> A layout ----
#pragma unroll
        for (int n = 0; n < 4; ++n)
#pragma unroll
            for (int r = 0; r < 8; ++r)
                Pw[(half * 8 + r) * LDH + n * 16 + ln] = (_Float16)S[n][r];

        // ---- O += P V ----
#pragma unroll
        for (int kk = 0; kk < 2; ++kk) {
            const v16h a = load_a16x32(&Pw[kk * 32], LDH, lane);
#pragma unroll
            for (int n = 0; n < 4; ++n) {
                const v16h b = load_b32x16(&Vt[cur][(n * 16) * LDH + kk * 32], LDH, lane);
                O[n] = wmma_f16(a, b, O[n]);
            }
        }

        // ---- transpose-store next V tile into spare buffer ----
        if (hasNext) {
#pragma unroll
            for (int j = 0; j < 2; ++j) {
                const int i = tid + 256 * j;
                const int r = i >> 3;
                const int c = (i & 7) << 3;
#pragma unroll
                for (int e = 0; e < 8; ++e) Vt[nxt][(c + e) * LDH + r] = vr[j].h[e];
            }
        }

        wait_async0();          // K(kt+1) resident in LDS
        __syncthreads();        // all waves done with cur, nxt fully staged
        cur = nxt;
    }

    // ---- normalize and store: out[b][srow][h*64 + col], f32 ----
    const int b = bh >> 4;
    const int h = bh & 15;
#pragma unroll
    for (int n = 0; n < 4; ++n)
#pragma unroll
        for (int r = 0; r < 8; ++r) {
            const int srow = qtile * 128 + wave * 16 + half * 8 + r;
            const float inv = 1.0f / l[r];
            out[((size_t)b * SEQ + srow) * (HEADS * D_OUT) + h * D_OUT + n * 16 + ln]
                = O[n][r] * inv;
        }
}

// ---------------------------------------------------------------------------
extern "C" void kernel_launch(void* const* d_in, const int* in_sizes, int n_in,
                              void* d_out, int out_size, void* d_ws, size_t ws_size,
                              hipStream_t stream) {
    const float* x  = (const float*)d_in[0];
    const float* Wq = (const float*)d_in[1];
    const float* Wk = (const float*)d_in[2];
    const float* Wv = (const float*)d_in[3];
    float* out = (float*)d_out;
    _Float16* qkv = (_Float16*)d_ws;   // 3 planes * 16 MB f16

    dim3 gProj(ROWS / 128, HEADS, 3);
    qkv_proj_kernel<<<gProj, 256, 0, stream>>>(x, Wq, Wk, Wv, qkv);

    dim3 gAttn(SEQ / 128, BATCH * HEADS);
    attn_kernel<<<gAttn, 256, 0, stream>>>(qkv, out);
}
